// HeteroGCNWithAllAttention_14018773254235
// MI455X (gfx1250) — compile-verified, matrix operand order checked by reading, not run on hardware
//
#include <hip/hip_runtime.h>
#include <hip/hip_bf16.h>

typedef __attribute__((ext_vector_type(16))) __bf16 v16bf;
typedef __attribute__((ext_vector_type(8)))  float  v8f;

#define NEG_SLOPE 0.2f

// ---------------- sizes (match reference) ----------------
#define N_AC   40000
#define N_WD   20000
#define D_IN   128
#define HEADS  4
#define DH     64
#define D_OUT  64
#define E_TIC  400000
#define E_W    200000
#define E_REL  400000

static inline int cdiv(long a, long b) { return (int)((a + b - 1) / b); }

// ---------------- elementwise helpers ----------------
__global__ void k_fill(float* p, float v, int n) {
  int i = blockIdx.x * blockDim.x + threadIdx.x;
  if (i < n) p[i] = v;
}

__global__ void k_relu_ip(float* p, int n) {
  int i = blockIdx.x * blockDim.x + threadIdx.x;
  if (i < n) p[i] = fmaxf(p[i], 0.f);
}

__global__ void k_mean_relu_ip(float* a, const float* b, int n) {
  int i = blockIdx.x * blockDim.x + threadIdx.x;
  if (i < n) a[i] = fmaxf((a[i] + b[i]) * 0.5f, 0.f);
}

__global__ void k_mean2(const float* a, const float* b, float* o, int n) {
  int i = blockIdx.x * blockDim.x + threadIdx.x;
  if (i < n) o[i] = (a[i] + b[i]) * 0.5f;
}

__global__ void k_bias_act(float* out, const float* bias, int total, int HD, int elu) {
  int i = blockIdx.x * blockDim.x + threadIdx.x;
  if (i >= total) return;
  float v = out[i] + bias[i % HD];
  if (elu) v = (v > 0.f) ? v : (expf(v) - 1.f);
  out[i] = v;
}

// ---------------- bf16 packing into WMMA register layout ----------------
// Packed A layout: [M/16][K/32][lane 0..31][16 bf16]
//   lane = half*16 + r16 ; element e<8 -> K = kt*32 + half*8 + e
//                          element e>=8 -> K = kt*32 + half*8 + (e-8) + 16
__global__ void k_pack_a(const float* __restrict__ A, __bf16* __restrict__ Ap, int M, int K) {
  int i = blockIdx.x * blockDim.x + threadIdx.x;
  if (i >= M * K) return;
  int e = i & 15;
  int t = i >> 4;
  int lane = t & 31;
  t >>= 5;
  int KT = K >> 5;
  int kt = t % KT;
  int tm = t / KT;
  int r16 = lane & 15, half = lane >> 4;
  int row = tm * 16 + r16;
  int k = kt * 32 + half * 8 + (e < 8 ? e : e + 8);
  Ap[i] = (__bf16)A[(size_t)row * K + k];
}

// Packed B layout: [K/32][N/16][lane 0..31][16 bf16]; lane -> column tn*16 + r16
__global__ void k_pack_b(const float* __restrict__ B, __bf16* __restrict__ Bp, int K, int N) {
  int i = blockIdx.x * blockDim.x + threadIdx.x;
  if (i >= K * N) return;
  int e = i & 15;
  int t = i >> 4;
  int lane = t & 31;
  t >>= 5;
  int NT = N >> 4;
  int tn = t % NT;
  int kt = t / NT;
  int r16 = lane & 15, half = lane >> 4;
  int col = tn * 16 + r16;
  int k = kt * 32 + half * 8 + (e < 8 ? e : e + 8);
  Bp[i] = (__bf16)B[(size_t)k * N + col];
}

// ---------------- CDNA5 async global->LDS copy helpers ----------------
__device__ __forceinline__ void async_ld_b128(unsigned lds_off, unsigned long long gaddr) {
  asm volatile("global_load_async_to_lds_b128 %0, %1, off"
               :: "v"(lds_off), "v"(gaddr) : "memory");
}
__device__ __forceinline__ void wait_async0() {
  asm volatile("s_wait_asynccnt 0x0" ::: "memory");
}

// ---------------- WMMA GEMM on pre-packed bf16, B staged in LDS ----------------
// C[M,N] = A[M,K] @ B[K,N]; one wave owns a 16-row strip and sweeps all N/16 tiles.
// Whole packed B (K*N bf16) loaded once per block into LDS via async-LDS path.
template <int K, int N>
__global__ void k_gemm_wmma_packed(const __bf16* __restrict__ Ap, const __bf16* __restrict__ Bp,
                                   float* __restrict__ C, int M) {
  constexpr int KT = K / 32;
  constexpr int NT = N / 16;
  __shared__ __bf16 sB[KT * NT * 512];   // <=64KB (well under 320KB WGP LDS)

  const int lane = threadIdx.x & 31;
  const int wave = threadIdx.x >> 5;

  // cooperative async copy of full packed B into LDS (16B per lane per iter)
  {
    const int bytes = KT * NT * 512 * 2;
    const char* gB = (const char*)Bp;
    for (int off = threadIdx.x * 16; off < bytes; off += blockDim.x * 16) {
      unsigned lds = (unsigned)(size_t)((char*)sB + off);
      async_ld_b128(lds, (unsigned long long)(size_t)(gB + off));
    }
    wait_async0();
    __syncthreads();
  }

  const int strips = M >> 4;
  const int tm = blockIdx.x * (blockDim.x >> 5) + wave;   // wave-uniform
  if (tm >= strips) return;                               // whole-wave exit

  // hold the full 16xK A strip in registers: KT x v16bf per lane
  v16bf a[KT];
  const v16bf* Av = (const v16bf*)Ap;
#pragma unroll
  for (int kt = 0; kt < KT; ++kt)
    a[kt] = Av[(size_t)(tm * KT + kt) * 32 + lane];

  const v16bf* sBv = (const v16bf*)sB;
  const int r16 = lane & 15, half = lane >> 4;

  for (int tn = 0; tn < NT; ++tn) {
    v8f acc = {};
#pragma unroll
    for (int kt = 0; kt < KT; ++kt) {
      v16bf b = sBv[(kt * NT + tn) * 32 + lane];
      acc = __builtin_amdgcn_wmma_f32_16x16x32_bf16(
          false, a[kt], false, b, (short)0, acc, false, false);
    }
    float* crow = C + (size_t)(tm * 16 + half * 8) * N + tn * 16 + r16;
#pragma unroll
    for (int r = 0; r < 8; ++r)
      crow[(size_t)r * N] = acc[r];
  }
}

// ---------------- GAT pieces ----------------
__global__ void k_node_att(const float* __restrict__ z, const float* __restrict__ avec,
                           float* __restrict__ out, int Nn, int Hh, int Dd) {
  int i = blockIdx.x * blockDim.x + threadIdx.x;
  if (i >= Nn * Hh) return;
  int n = i / Hh, h = i - n * Hh;
  const float* zp = z + (size_t)n * Hh * Dd + (size_t)h * Dd;
  const float* ap = avec + (size_t)h * Dd;
  float s = 0.f;
  for (int d = 0; d < Dd; ++d) s += zp[d] * ap[d];
  out[i] = s;
}

__device__ inline void atomicMaxF(float* addr, float val) {
  unsigned int* ai = (unsigned int*)addr;
  unsigned int old = __float_as_uint(*addr);
  while (__uint_as_float(old) < val) {
    unsigned int assumed = old;
    old = atomicCAS(ai, assumed, __float_as_uint(val));
    if (old == assumed) break;
  }
}

__global__ void k_edge_logit(const int* __restrict__ src, const int* __restrict__ dst,
                             const float* __restrict__ el, const float* __restrict__ er,
                             float* __restrict__ ebuf, float* __restrict__ m, int E, int Hh) {
  int i = blockIdx.x * blockDim.x + threadIdx.x;
  if (i >= E * Hh) return;
  int eid = i / Hh, h = i - eid * Hh;
  float e = el[src[eid] * Hh + h] + er[dst[eid] * Hh + h];
  e = (e > 0.f) ? e : NEG_SLOPE * e;
  ebuf[i] = e;
  atomicMaxF(&m[dst[eid] * Hh + h], e);
}

__global__ void k_edge_exp(const int* __restrict__ dst, float* __restrict__ ebuf,
                           const float* __restrict__ m, float* __restrict__ s, int E, int Hh) {
  int i = blockIdx.x * blockDim.x + threadIdx.x;
  if (i >= E * Hh) return;
  int eid = i / Hh, h = i - eid * Hh;
  float ex = expf(ebuf[i] - m[dst[eid] * Hh + h]);
  ebuf[i] = ex;
  atomicAdd(&s[dst[eid] * Hh + h], ex);
}

__global__ void k_edge_coef(const int* __restrict__ dst, const float* __restrict__ ew,
                            float* __restrict__ ebuf, const float* __restrict__ s, int E, int Hh) {
  int i = blockIdx.x * blockDim.x + threadIdx.x;
  if (i >= E * Hh) return;
  int eid = i / Hh, h = i - eid * Hh;
  ebuf[i] = ebuf[i] / (s[dst[eid] * Hh + h] + 1e-9f) * ew[eid];
}

__global__ void k_edge_agg(const int* __restrict__ src, const int* __restrict__ dst,
                           const float* __restrict__ coef, const float* __restrict__ z_src,
                           float* __restrict__ out, long total, int Hh, int Dd) {
  long i = (long)blockIdx.x * blockDim.x + threadIdx.x;
  if (i >= total) return;
  int d   = (int)(i % Dd);
  long t  = i / Dd;
  int h   = (int)(t % Hh);
  int eid = (int)(t / Hh);
  float v = z_src[(size_t)src[eid] * Hh * Dd + (size_t)h * Dd + d] * coef[eid * Hh + h];
  atomicAdd(&out[(size_t)dst[eid] * Hh * Dd + (size_t)h * Dd + d], v);
}

// ---------------- host-side orchestration ----------------
static void run_gat(const float* z_src, const float* z_dst,
                    const int* src, const int* dst, const float* ew,
                    const float* al, const float* ar, const float* bias,
                    float* out_acc,
                    float* el, float* er, float* m, float* sbuf, float* ecoef,
                    int Ns, int Nd, int E, int Hh, int Dd, int elu, hipStream_t s) {
  const int T = 256;
  int HD = Hh * Dd;
  k_fill<<<cdiv((long)Nd * HD, T), T, 0, s>>>(out_acc, 0.f, Nd * HD);
  k_fill<<<cdiv((long)Nd * Hh, T), T, 0, s>>>(m, -1e30f, Nd * Hh);
  k_fill<<<cdiv((long)Nd * Hh, T), T, 0, s>>>(sbuf, 0.f, Nd * Hh);
  k_node_att<<<cdiv((long)Ns * Hh, T), T, 0, s>>>(z_src, al, el, Ns, Hh, Dd);
  k_node_att<<<cdiv((long)Nd * Hh, T), T, 0, s>>>(z_dst, ar, er, Nd, Hh, Dd);
  k_edge_logit<<<cdiv((long)E * Hh, T), T, 0, s>>>(src, dst, el, er, ecoef, m, E, Hh);
  k_edge_exp  <<<cdiv((long)E * Hh, T), T, 0, s>>>(dst, ecoef, m, sbuf, E, Hh);
  k_edge_coef <<<cdiv((long)E * Hh, T), T, 0, s>>>(dst, ew, ecoef, sbuf, E, Hh);
  long tot = (long)E * Hh * Dd;
  k_edge_agg<<<cdiv(tot, T), T, 0, s>>>(src, dst, ecoef, z_src, out_acc, tot, Hh, Dd);
  k_bias_act<<<cdiv((long)Nd * HD, T), T, 0, s>>>(out_acc, bias, Nd * HD, HD, elu);
}

// ---------- workspace layout: float region (element offsets) ----------
#define WS_Z_TIC    0L
#define WS_Z_W      10240000L
#define WS_Z_RELS   15360000L
#define WS_Z_RELD   25600000L
#define WS_H_TIC    30720000L
#define WS_H_W      40960000L
#define WS_H_REL    46080000L
#define WS_Z2_TIC   51200000L
#define WS_Z2_W     53760000L
#define WS_Z2_RELS  55040000L
#define WS_Z2_RELD  57600000L
#define WS_O_W      58880000L
#define WS_O_REL    60160000L
#define WS_EL       61440000L
#define WS_ER       61600000L
#define WS_M        61760000L
#define WS_S        61920000L
#define WS_ECOEF    62080000L
// ---------- bf16 region (byte offsets from start of ws) ----------
#define WB_PA_XAC   254720000L   // 40000*128 bf16
#define WB_PA_XW    264960000L   // 20000*128 bf16
#define WB_PA_HTIC  270080000L   // 40000*256 bf16
#define WB_PA_HW    290560000L   // 20000*256 bf16
#define WB_PB_W1TIC 300800000L   // 128*256 bf16
#define WB_PB_W1W   300865536L
#define WB_PB_W1REL 300931072L
#define WB_PB_W2TIC 300996608L   // 256*64 bf16
#define WB_PB_W2W   301029376L
#define WB_PB_W2REL 301062144L

extern "C" void kernel_launch(void* const* d_in, const int* in_sizes, int n_in,
                              void* d_out, int out_size, void* d_ws, size_t ws_size,
                              hipStream_t stream) {
  const float* x_ac  = (const float*)d_in[0];
  const float* x_w   = (const float*)d_in[1];
  const float* w_tic = (const float*)d_in[2];
  const float* w_w   = (const float*)d_in[3];
  const float* w_rel = (const float*)d_in[4];
  const float* W1_tic = (const float*)d_in[5];
  const float* al1_tic = (const float*)d_in[6];
  const float* ar1_tic = (const float*)d_in[7];
  const float* b1_tic = (const float*)d_in[8];
  const float* W1_w = (const float*)d_in[9];
  const float* al1_w = (const float*)d_in[10];
  const float* ar1_w = (const float*)d_in[11];
  const float* b1_w = (const float*)d_in[12];
  const float* W1_rel = (const float*)d_in[13];
  const float* al1_rel = (const float*)d_in[14];
  const float* ar1_rel = (const float*)d_in[15];
  const float* b1_rel = (const float*)d_in[16];
  const float* W2_tic = (const float*)d_in[17];
  const float* al2_tic = (const float*)d_in[18];
  const float* ar2_tic = (const float*)d_in[19];
  const float* b2_tic = (const float*)d_in[20];
  const float* W2_w = (const float*)d_in[21];
  const float* al2_w = (const float*)d_in[22];
  const float* ar2_w = (const float*)d_in[23];
  const float* b2_w = (const float*)d_in[24];
  const float* W2_rel = (const float*)d_in[25];
  const float* al2_rel = (const float*)d_in[26];
  const float* ar2_rel = (const float*)d_in[27];
  const float* b2_rel = (const float*)d_in[28];
  const int* src_tic = (const int*)d_in[29];
  const int* dst_tic = (const int*)d_in[30];
  const int* src_w   = (const int*)d_in[31];
  const int* dst_w   = (const int*)d_in[32];
  const int* src_rel = (const int*)d_in[33];
  const int* dst_rel = (const int*)d_in[34];

  float* ws  = (float*)d_ws;
  char*  wsb = (char*)d_ws;
  float* out = (float*)d_out;
  float* el = ws + WS_EL; float* er = ws + WS_ER;
  float* m  = ws + WS_M;  float* sb = ws + WS_S;  float* ec = ws + WS_ECOEF;

  __bf16* paXac  = (__bf16*)(wsb + WB_PA_XAC);
  __bf16* paXw   = (__bf16*)(wsb + WB_PA_XW);
  __bf16* paHtic = (__bf16*)(wsb + WB_PA_HTIC);
  __bf16* paHw   = (__bf16*)(wsb + WB_PA_HW);
  __bf16* pbW1t  = (__bf16*)(wsb + WB_PB_W1TIC);
  __bf16* pbW1w  = (__bf16*)(wsb + WB_PB_W1W);
  __bf16* pbW1r  = (__bf16*)(wsb + WB_PB_W1REL);
  __bf16* pbW2t  = (__bf16*)(wsb + WB_PB_W2TIC);
  __bf16* pbW2w  = (__bf16*)(wsb + WB_PB_W2W);
  __bf16* pbW2r  = (__bf16*)(wsb + WB_PB_W2REL);

  const int T = 256;

  // -------- pack weights (bf16, WMMA-B layout) --------
  k_pack_b<<<cdiv(D_IN * 256, T), T, 0, stream>>>(W1_tic, pbW1t, D_IN, HEADS * DH);
  k_pack_b<<<cdiv(D_IN * 256, T), T, 0, stream>>>(W1_w,   pbW1w, D_IN, HEADS * DH);
  k_pack_b<<<cdiv(D_IN * 256, T), T, 0, stream>>>(W1_rel, pbW1r, D_IN, HEADS * DH);
  k_pack_b<<<cdiv(256 * D_OUT, T), T, 0, stream>>>(W2_tic, pbW2t, HEADS * DH, D_OUT);
  k_pack_b<<<cdiv(256 * D_OUT, T), T, 0, stream>>>(W2_w,   pbW2w, HEADS * DH, D_OUT);
  k_pack_b<<<cdiv(256 * D_OUT, T), T, 0, stream>>>(W2_rel, pbW2r, HEADS * DH, D_OUT);

  // -------- pack layer-1 activations (bf16, WMMA-A layout) --------
  k_pack_a<<<cdiv((long)N_AC * D_IN, T), T, 0, stream>>>(x_ac, paXac, N_AC, D_IN);
  k_pack_a<<<cdiv((long)N_WD * D_IN, T), T, 0, stream>>>(x_w,  paXw,  N_WD, D_IN);

  // -------- layer-1 projections (WMMA bf16, B in LDS via async copy) --------
  k_gemm_wmma_packed<128, 256><<<cdiv(N_AC / 16, 8), 256, 0, stream>>>(paXac, pbW1t, ws + WS_Z_TIC,  N_AC);
  k_gemm_wmma_packed<128, 256><<<cdiv(N_WD / 16, 8), 256, 0, stream>>>(paXw,  pbW1w, ws + WS_Z_W,    N_WD);
  k_gemm_wmma_packed<128, 256><<<cdiv(N_AC / 16, 8), 256, 0, stream>>>(paXac, pbW1r, ws + WS_Z_RELS, N_AC);
  k_gemm_wmma_packed<128, 256><<<cdiv(N_WD / 16, 8), 256, 0, stream>>>(paXw,  pbW1r, ws + WS_Z_RELD, N_WD);

  // -------- layer-1 GATs (ELU inside conv) --------
  run_gat(ws + WS_Z_TIC, ws + WS_Z_TIC, src_tic, dst_tic, w_tic,
          al1_tic, ar1_tic, b1_tic, ws + WS_H_TIC,
          el, er, m, sb, ec, N_AC, N_AC, E_TIC, HEADS, DH, 1, stream);
  run_gat(ws + WS_Z_W, ws + WS_Z_W, src_w, dst_w, w_w,
          al1_w, ar1_w, b1_w, ws + WS_H_W,
          el, er, m, sb, ec, N_WD, N_WD, E_W, HEADS, DH, 1, stream);
  run_gat(ws + WS_Z_RELS, ws + WS_Z_RELD, src_rel, dst_rel, w_rel,
          al1_rel, ar1_rel, b1_rel, ws + WS_H_REL,
          el, er, m, sb, ec, N_AC, N_WD, E_REL, HEADS, DH, 1, stream);

  // HeteroGraphConv mean over etypes, flatten heads, ReLU
  k_relu_ip<<<cdiv((long)N_AC * HEADS * DH, T), T, 0, stream>>>(ws + WS_H_TIC, N_AC * HEADS * DH);
  k_mean_relu_ip<<<cdiv((long)N_WD * HEADS * DH, T), T, 0, stream>>>(
      ws + WS_H_W, ws + WS_H_REL, N_WD * HEADS * DH);

  // -------- pack layer-2 activations --------
  k_pack_a<<<cdiv((long)N_AC * 256, T), T, 0, stream>>>(ws + WS_H_TIC, paHtic, N_AC, HEADS * DH);
  k_pack_a<<<cdiv((long)N_WD * 256, T), T, 0, stream>>>(ws + WS_H_W,   paHw,   N_WD, HEADS * DH);

  // -------- layer-2 projections (WMMA bf16) --------
  k_gemm_wmma_packed<256, 64><<<cdiv(N_AC / 16, 8), 256, 0, stream>>>(paHtic, pbW2t, ws + WS_Z2_TIC,  N_AC);
  k_gemm_wmma_packed<256, 64><<<cdiv(N_WD / 16, 8), 256, 0, stream>>>(paHw,   pbW2w, ws + WS_Z2_W,    N_WD);
  k_gemm_wmma_packed<256, 64><<<cdiv(N_AC / 16, 8), 256, 0, stream>>>(paHtic, pbW2r, ws + WS_Z2_RELS, N_AC);
  k_gemm_wmma_packed<256, 64><<<cdiv(N_WD / 16, 8), 256, 0, stream>>>(paHw,   pbW2r, ws + WS_Z2_RELD, N_WD);

  // -------- layer-2 GATs (1 head, no activation) --------
  run_gat(ws + WS_Z2_TIC, ws + WS_Z2_TIC, src_tic, dst_tic, w_tic,
          al2_tic, ar2_tic, b2_tic, out /* o_tic directly */,
          el, er, m, sb, ec, N_AC, N_AC, E_TIC, 1, D_OUT, 0, stream);
  run_gat(ws + WS_Z2_W, ws + WS_Z2_W, src_w, dst_w, w_w,
          al2_w, ar2_w, b2_w, ws + WS_O_W,
          el, er, m, sb, ec, N_WD, N_WD, E_W, 1, D_OUT, 0, stream);
  run_gat(ws + WS_Z2_RELS, ws + WS_Z2_RELD, src_rel, dst_rel, w_rel,
          al2_rel, ar2_rel, b2_rel, ws + WS_O_REL,
          el, er, m, sb, ec, N_AC, N_WD, E_REL, 1, D_OUT, 0, stream);

  // o_word = (o_w + o_rel) * 0.5
  k_mean2<<<cdiv((long)N_WD * D_OUT, T), T, 0, stream>>>(
      ws + WS_O_W, ws + WS_O_REL, out + (long)N_AC * D_OUT, N_WD * D_OUT);
}
